// ResonanceAttention_31653908972324
// MI455X (gfx1250) — matrix operand order repaired and checked by reading, not compiled, not run on hardware
//
#include <hip/hip_runtime.h>
#include <hip/hip_bf16.h>

#define HIDDEN   1024
#define HEADS    16
#define HEAD_DIM 64
#define SEQ      2048
#define BATCH    2
#define MROWS    (BATCH * SEQ)          // 4096
#define PRIME_SCALE 0.047245559f        // 1/sqrt(7*64)
#define ENT_THRESH 0.2f
#define EPS_F 1e-6f

typedef __attribute__((ext_vector_type(16))) __bf16 v16bf;
typedef __attribute__((ext_vector_type(8)))  float  v8f;

union FragAB {
    v16bf v;
    unsigned int u32[8];
};

#define WMMA_BF16(A, B, C) \
    __builtin_amdgcn_wmma_f32_16x16x32_bf16(false, (A), false, (B), (short)0, (C), false, false)

__device__ __forceinline__ unsigned short f2bf(float f) {
    union { float f; unsigned int u; } x; x.f = f;
    unsigned int r = x.u + 0x7fffu + ((x.u >> 16) & 1u);   // round-to-nearest-even
    return (unsigned short)(r >> 16);
}

// ---------------------------------------------------------------------------
// Async global->LDS 16-byte copy (GLOBAL_LOAD_ASYNC_TO_LDS_B128, ASYNCcnt).
// LDS address = low 32 bits of the generic pointer (ISA 10.2 truncation).
// Async loads complete in order, so waiting asynccnt<=N after issuing N new
// ops guarantees the previous group has landed (ping-pong double buffering).
// ---------------------------------------------------------------------------
__device__ __forceinline__ void async_copy_b128(void* lds_dst, const void* gsrc) {
    unsigned int l = (unsigned int)(unsigned long long)lds_dst;
    asm volatile("global_load_async_to_lds_b128 %0, %1, off"
                 :: "v"(l), "v"(gsrc)
                 : "memory");
}
__device__ __forceinline__ void async_wait0() { asm volatile("s_wait_asynccnt 0x0" ::: "memory"); }
__device__ __forceinline__ void async_wait3() { asm volatile("s_wait_asynccnt 0x3" ::: "memory"); }
__device__ __forceinline__ void async_wait4() { asm volatile("s_wait_asynccnt 0x4" ::: "memory"); }

// A fragment: 16x32 bf16 tile from LDS, row-major [row][k], leading dim ld.
__device__ __forceinline__ void load_a_frag(FragAB& a, const unsigned short* lds,
                                            int ld, int mbase, int kof, int lane) {
    const unsigned short* p = lds + (mbase + (lane & 15)) * ld + kof;
    const int kb = (lane >> 4) << 4;   // 0 or 16
#pragma unroll
    for (int i = 0; i < 4; ++i)
        a.u32[i] = *(const unsigned int*)(p + kb + 2 * i);
#pragma unroll
    for (int i = 0; i < 4; ++i)
        a.u32[4 + i] = *(const unsigned int*)(p + 8 + kb + 2 * i);
}

// B fragment: 32x16 bf16, stored transposed in LDS as [n][k] (ld over k).
__device__ __forceinline__ void load_b_frag(FragAB& b, const unsigned short* lds,
                                            int ld, int nbase, int kof, int lane) {
    const unsigned short* p = lds + (nbase + (lane & 15)) * ld + kof + ((lane >> 4) << 4);
#pragma unroll
    for (int i = 0; i < 8; ++i)
        b.u32[i] = *(const unsigned int*)(p + 2 * i);
}

__global__ void zero_scalar_kernel(float* p) { *p = 0.f; }

// ---------------------------------------------------------------------------
// One-shot fp32 -> bf16 conversion (memory bound pre-pass).  n % 2048 == 0.
// ---------------------------------------------------------------------------
__global__ __launch_bounds__(256) void cvt_f32_bf16_kernel(
    const float* __restrict__ s, unsigned short* __restrict__ d, int n)
{
    const int i = (blockIdx.x * 256 + threadIdx.x) * 8;
    if (i >= n) return;
    float4 a = *(const float4*)(s + i);
    float4 b = *(const float4*)(s + i + 4);
    uint4 o;
    o.x = (unsigned)f2bf(a.x) | ((unsigned)f2bf(a.y) << 16);
    o.y = (unsigned)f2bf(a.z) | ((unsigned)f2bf(a.w) << 16);
    o.z = (unsigned)f2bf(b.x) | ((unsigned)f2bf(b.y) << 16);
    o.w = (unsigned)f2bf(b.z) | ((unsigned)f2bf(b.w) << 16);
    *(uint4*)(d + i) = o;
}

// ---------------------------------------------------------------------------
// QKV projection from bf16 X / bf16 W:  out = (X @ W^T + b) [*scale]
// Block tile 128x64, K-step 32, double-buffered async LDS staging.
// Each of 8 waves computes a 32x32 quadrant (2 A-frags x 2 B-frags, 4 WMMA).
// q,k stored [B,H,S,D]; v stored transposed [B,H,D,S].
// grid = (32, 16, 3), block = 256.
// ---------------------------------------------------------------------------
__global__ __launch_bounds__(256) void qkv_gemm_kernel(
    const unsigned short* __restrict__ X,
    const unsigned short* __restrict__ Wq, const float* __restrict__ bq,
    const unsigned short* __restrict__ Wk, const float* __restrict__ bk,
    const unsigned short* __restrict__ Wv, const float* __restrict__ bv,
    unsigned short* __restrict__ qo, unsigned short* __restrict__ ko,
    unsigned short* __restrict__ vo)
{
    __shared__ unsigned short As[2][128 * 32];
    __shared__ unsigned short Bs[2][64 * 32];

    const unsigned short* W; const float* bias; unsigned short* outp; float scl;
    if (blockIdx.z == 0)      { W = Wq; bias = bq; outp = qo; scl = PRIME_SCALE; }
    else if (blockIdx.z == 1) { W = Wk; bias = bk; outp = ko; scl = 1.f; }
    else                      { W = Wv; bias = bv; outp = vo; scl = 1.f; }

    const int m0 = blockIdx.x * 128;
    const int n0 = blockIdx.y * 64;
    const int t = threadIdx.x;
    const int wave = t >> 5, lane = t & 31;
    const int mbase = (wave & 3) * 32;
    const int nbase = (wave >> 2) * 32;
    const int hoff = (lane & 16) ? 8 : 0;

    const int rowA0 = t >> 2, rowA1 = rowA0 + 64;   // 128 rows, 4 chunks each
    const int offA  = (t & 3) * 8;
    const int rowB  = t >> 2;                       // 64 rows
    const int offB  = (t & 3) * 8;

    auto stage = [&](int buf, int k0) {
        async_copy_b128(&As[buf][rowA0 * 32 + offA], X + (size_t)(m0 + rowA0) * HIDDEN + k0 + offA);
        async_copy_b128(&As[buf][rowA1 * 32 + offA], X + (size_t)(m0 + rowA1) * HIDDEN + k0 + offA);
        async_copy_b128(&Bs[buf][rowB * 32 + offB],  W + (size_t)(n0 + rowB) * HIDDEN + k0 + offB);
    };

    v8f acc[2][2] = {};

    stage(0, 0);
    for (int kt = 0; kt < HIDDEN / 32; ++kt) {
        const int cur = kt & 1;
        if (kt + 1 < HIDDEN / 32) { stage(cur ^ 1, (kt + 1) * 32); async_wait3(); }
        else                      { async_wait0(); }
        __syncthreads();
        FragAB a0, a1, b0, b1;
        load_a_frag(a0, As[cur], 32, mbase,      0, lane);
        load_a_frag(a1, As[cur], 32, mbase + 16, 0, lane);
        load_b_frag(b0, Bs[cur], 32, nbase,      0, lane);
        load_b_frag(b1, Bs[cur], 32, nbase + 16, 0, lane);
        acc[0][0] = WMMA_BF16(a0.v, b0.v, acc[0][0]);
        acc[0][1] = WMMA_BF16(a0.v, b1.v, acc[0][1]);
        acc[1][0] = WMMA_BF16(a1.v, b0.v, acc[1][0]);
        acc[1][1] = WMMA_BF16(a1.v, b1.v, acc[1][1]);
        __syncthreads();
    }

#pragma unroll
    for (int jm = 0; jm < 2; ++jm) {
#pragma unroll
        for (int jn = 0; jn < 2; ++jn) {
            const int col = n0 + nbase + jn * 16 + (lane & 15);
            const float bvv = bias[col];
            const int h = col >> 6, d = col & 63;
#pragma unroll
            for (int i = 0; i < 8; ++i) {
                const int row = m0 + mbase + jm * 16 + hoff + i;
                const int bidx = row >> 11, s = row & (SEQ - 1);
                const float val = (acc[jm][jn][i] + bvv) * scl;
                if (blockIdx.z == 2)   // V: [B,H,D,S]
                    outp[((size_t)((bidx * HEADS + h) * HEAD_DIM + d)) * SEQ + s] = f2bf(val);
                else                   // Q,K: [B,H,S,D]
                    outp[((size_t)((bidx * HEADS + h) * SEQ + s) << 6) + d] = f2bf(val);
            }
        }
    }
}

// ---------------------------------------------------------------------------
// Flash attention per (qblock, b*H+h): online softmax + streamed entropy,
// double-buffered async K/V staging.  grid = (32, 32), block = 256 (8 waves).
// ---------------------------------------------------------------------------
__global__ __launch_bounds__(256) void attn_kernel(
    const unsigned short* __restrict__ Q,   // [B,H,S,D]
    const unsigned short* __restrict__ K,   // [B,H,S,D]
    const unsigned short* __restrict__ V,   // [B,H,D,S]
    const int* __restrict__ mask,
    unsigned short* __restrict__ merged,    // [B,S,HIDDEN]
    float* __restrict__ Hsum)
{
    __shared__ unsigned short Qs[64 * 64];
    __shared__ unsigned short Ks[2][64 * 64];
    __shared__ unsigned short Vts[2][64 * 64];   // [d][key]
    __shared__ unsigned short Ps[64 * 64];       // exp(s - m), bf16
    __shared__ float Ss[64 * 64];
    __shared__ float rowScale[64];
    __shared__ float rowInvL[64];
    __shared__ float redH[64];

    const int qb = blockIdx.x;
    const int bh = blockIdx.y;
    const int b = bh >> 4, h = bh & 15;
    const int t = threadIdx.x;
    const int wave = t >> 5, lane = t & 31;
    const int mbase = (wave & 3) * 16;
    const int dbase = (wave >> 2) * 32;
    const int hoff = (lane & 16) ? 8 : 0;
    const int q0 = qb * 64;
    const size_t headOff = ((size_t)(b * HEADS + h)) * SEQ * HEAD_DIM;

    const int r0 = t >> 3,         o0 = (t & 7) * 8;          // chunk t
    const int r1 = (t + 256) >> 3, o1 = ((t + 256) & 7) * 8;  // chunk t+256

    auto stageKV = [&](int buf, int key0) {
        async_copy_b128(&Ks[buf][r0 * 64 + o0],  K + headOff + (size_t)(key0 + r0) * 64 + o0);
        async_copy_b128(&Ks[buf][r1 * 64 + o1],  K + headOff + (size_t)(key0 + r1) * 64 + o1);
        async_copy_b128(&Vts[buf][r0 * 64 + o0], V + headOff + (size_t)r0 * SEQ + key0 + o0);
        async_copy_b128(&Vts[buf][r1 * 64 + o1], V + headOff + (size_t)r1 * SEQ + key0 + o1);
    };

    // stage Q tile (ordered before first K/V group; covered by first wait)
    async_copy_b128(&Qs[r0 * 64 + o0], Q + headOff + (size_t)(q0 + r0) * 64 + o0);
    async_copy_b128(&Qs[r1 * 64 + o1], Q + headOff + (size_t)(q0 + r1) * 64 + o1);
    stageKV(0, 0);

    float rm = -3.0e38f, rl = 0.f, rT = 0.f;
    int nvalid = 0;
    v8f acc0 = {}; v8f acc1 = {};

    for (int kb = 0; kb < SEQ / 64; ++kb) {
        const int cur = kb & 1;
        const int key0 = kb * 64;
        if (kb + 1 < SEQ / 64) { stageKV(cur ^ 1, (kb + 1) * 64); async_wait4(); }
        else                   { async_wait0(); }
        __syncthreads();

        // scores = Q K^T (fp32 accum) for this wave's 16x32 patch
        v8f s0 = {}; v8f s1 = {};
#pragma unroll
        for (int kof = 0; kof < 64; kof += 32) {
            FragAB a, b0, b1;
            load_a_frag(a,  Qs, 64, mbase, kof, lane);
            load_b_frag(b0, Ks[cur], 64, dbase, kof, lane);
            load_b_frag(b1, Ks[cur], 64, dbase + 16, kof, lane);
            s0 = WMMA_BF16(a.v, b0.v, s0);
            s1 = WMMA_BF16(a.v, b1.v, s1);
        }
        // mask + spill scores to LDS
#pragma unroll
        for (int j = 0; j < 2; ++j) {
            v8f sv = j ? s1 : s0;
            const int col = dbase + j * 16 + (lane & 15);
            const bool ok = (mask[b * SEQ + key0 + col] != 0);
#pragma unroll
            for (int i = 0; i < 8; ++i)
                Ss[(mbase + hoff + i) * 64 + col] = ok ? sv[i] : -1.0e30f;
        }
        __syncthreads();

        // streaming softmax state per query row (thread t < 64 owns row t)
        if (t < 64) {
            float bm = -3.0e38f;
            for (int k = 0; k < 64; ++k) bm = fmaxf(bm, Ss[t * 64 + k]);
            const float m2 = fmaxf(rm, bm);
            const float sc = __expf(rm - m2);
            float lb = 0.f, Tb = 0.f;
            for (int k = 0; k < 64; ++k) {
                const float s = Ss[t * 64 + k];
                float e = 0.f;
                if (s > -1.0e29f) { e = __expf(s - m2); ++nvalid; }
                lb += e; Tb += e * s;
                Ps[t * 64 + k] = f2bf(e);
            }
            rl = rl * sc + lb;
            rT = rT * sc + Tb;
            rm = m2;
            rowScale[t] = sc;
        }
        __syncthreads();

        // rescale running accumulator and add P @ V
#pragma unroll
        for (int i = 0; i < 8; ++i) {
            const float sc = rowScale[mbase + hoff + i];
            acc0[i] *= sc; acc1[i] *= sc;
        }
#pragma unroll
        for (int kof = 0; kof < 64; kof += 32) {
            FragAB a, b0, b1;
            load_a_frag(a,  Ps, 64, mbase, kof, lane);
            load_b_frag(b0, Vts[cur], 64, dbase, kof, lane);
            load_b_frag(b1, Vts[cur], 64, dbase + 16, kof, lane);
            acc0 = WMMA_BF16(a.v, b0.v, acc0);
            acc1 = WMMA_BF16(a.v, b1.v, acc1);
        }
        __syncthreads();
    }

    if (t < 64) {
        rowInvL[t] = 1.f / rl;
        // H = -sum p*log(p+eps)  ~=  m + ln(l) - T/l - eps * n_valid
        redH[t] = rm + __logf(rl) - rT / rl - EPS_F * (float)nvalid;
    }
    __syncthreads();
    if (t == 0) {
        float s = 0.f;
        for (int i = 0; i < 64; ++i) s += redH[i];
        atomicAdd(Hsum, s);
    }

    // write normalized output, merged layout [B, S, HIDDEN]
#pragma unroll
    for (int j = 0; j < 2; ++j) {
        v8f acc = j ? acc1 : acc0;
        const int d = dbase + j * 16 + (lane & 15);
#pragma unroll
        for (int i = 0; i < 8; ++i) {
            const int qrow = mbase + hoff + i;
            const float val = acc[i] * rowInvL[qrow];
            merged[(size_t)(b * SEQ + q0 + qrow) * HIDDEN + h * 64 + d] = f2bf(val);
        }
    }
}

// ---------------------------------------------------------------------------
// Output GEMM: out = sel * (merged @ o_w^T) + o_b   (sel from global entropy)
// Block tile 128x64, double-buffered.  grid = (32, 16), block = 256.
// ---------------------------------------------------------------------------
__global__ __launch_bounds__(256) void out_gemm_kernel(
    const unsigned short* __restrict__ A,    // merged bf16 [4096][1024]
    const unsigned short* __restrict__ Wo,   // bf16 [1024][1024]
    const float* __restrict__ bo,
    const float* __restrict__ Hsum,
    float* __restrict__ out)
{
    __shared__ unsigned short As[2][128 * 32];
    __shared__ unsigned short Bs[2][64 * 32];

    const int m0 = blockIdx.x * 128;
    const int n0 = blockIdx.y * 64;
    const int t = threadIdx.x;
    const int wave = t >> 5, lane = t & 31;
    const int mbase = (wave & 3) * 32;
    const int nbase = (wave >> 2) * 32;
    const int hoff = (lane & 16) ? 8 : 0;

    const int rowA0 = t >> 2, rowA1 = rowA0 + 64;
    const int offA  = (t & 3) * 8;
    const int rowB  = t >> 2;
    const int offB  = (t & 3) * 8;

    // entropy-gated scalar (identical in every block; 10-step blend is constant)
    const float avgH = Hsum[0] / (float)(BATCH * HEADS * SEQ);
    float cfin = 1.f;
#pragma unroll
    for (int i = 1; i < 10; ++i) {
        const float bl = 1.f / (1.f + __expf(-(float)i * 0.1f));
        cfin = (1.f - bl) * cfin + bl * (0.5f + 0.5f * bl);
    }
    const float sel = (avgH < ENT_THRESH) ? 1.f : cfin;

    auto stage = [&](int buf, int k0) {
        async_copy_b128(&As[buf][rowA0 * 32 + offA], A  + (size_t)(m0 + rowA0) * HIDDEN + k0 + offA);
        async_copy_b128(&As[buf][rowA1 * 32 + offA], A  + (size_t)(m0 + rowA1) * HIDDEN + k0 + offA);
        async_copy_b128(&Bs[buf][rowB * 32 + offB],  Wo + (size_t)(n0 + rowB) * HIDDEN + k0 + offB);
    };

    v8f acc[2][2] = {};

    stage(0, 0);
    for (int kt = 0; kt < HIDDEN / 32; ++kt) {
        const int cur = kt & 1;
        if (kt + 1 < HIDDEN / 32) { stage(cur ^ 1, (kt + 1) * 32); async_wait3(); }
        else                      { async_wait0(); }
        __syncthreads();
        FragAB a0, a1, b0, b1;
        load_a_frag(a0, As[cur], 32, mbase,      0, lane);
        load_a_frag(a1, As[cur], 32, mbase + 16, 0, lane);
        load_b_frag(b0, Bs[cur], 32, nbase,      0, lane);
        load_b_frag(b1, Bs[cur], 32, nbase + 16, 0, lane);
        acc[0][0] = WMMA_BF16(a0.v, b0.v, acc[0][0]);
        acc[0][1] = WMMA_BF16(a0.v, b1.v, acc[0][1]);
        acc[1][0] = WMMA_BF16(a1.v, b0.v, acc[1][0]);
        acc[1][1] = WMMA_BF16(a1.v, b1.v, acc[1][1]);
        __syncthreads();
    }

#pragma unroll
    for (int jm = 0; jm < 2; ++jm) {
#pragma unroll
        for (int jn = 0; jn < 2; ++jn) {
            const int col = n0 + nbase + jn * 16 + (lane & 15);
            const float bvv = bo[col];
#pragma unroll
            for (int i = 0; i < 8; ++i) {
                const int row = m0 + mbase + jm * 16 + hoff + i;
                out[(size_t)row * HIDDEN + col] = sel * acc[jm][jn][i] + bvv;
            }
        }
    }
}

// ---------------------------------------------------------------------------
extern "C" void kernel_launch(void* const* d_in, const int* in_sizes, int n_in,
                              void* d_out, int out_size, void* d_ws, size_t ws_size,
                              hipStream_t stream)
{
    (void)in_sizes; (void)n_in; (void)out_size; (void)ws_size;

    const float* X   = (const float*)d_in[0];
    const float* qw  = (const float*)d_in[1];
    const float* qbi = (const float*)d_in[2];
    const float* kw  = (const float*)d_in[3];
    const float* kbi = (const float*)d_in[4];
    const float* vw  = (const float*)d_in[5];
    const float* vbi = (const float*)d_in[6];
    const float* ow  = (const float*)d_in[7];
    const float* obi = (const float*)d_in[8];
    const int*   msk = (const int*)d_in[9];
    float* out = (float*)d_out;

    char* ws = (char*)d_ws;
    const size_t E  = (size_t)MROWS * HIDDEN;            // 4M elements
    const size_t E2 = E * 2;                             // 8 MB
    const size_t W2 = (size_t)HIDDEN * HIDDEN * 2;       // 2 MB
    unsigned short* qg  = (unsigned short*)(ws + 0 * E2);
    unsigned short* kg  = (unsigned short*)(ws + 1 * E2);
    unsigned short* vg  = (unsigned short*)(ws + 2 * E2);
    unsigned short* mg  = (unsigned short*)(ws + 3 * E2);
    unsigned short* xb  = (unsigned short*)(ws + 4 * E2);
    unsigned short* wqb = (unsigned short*)(ws + 5 * E2 + 0 * W2);
    unsigned short* wkb = (unsigned short*)(ws + 5 * E2 + 1 * W2);
    unsigned short* wvb = (unsigned short*)(ws + 5 * E2 + 2 * W2);
    unsigned short* wob = (unsigned short*)(ws + 5 * E2 + 3 * W2);
    float* Hs           = (float*)(ws + 5 * E2 + 4 * W2);

    const int NW = HIDDEN * HIDDEN;                      // 1M
    zero_scalar_kernel<<<1, 1, 0, stream>>>(Hs);
    cvt_f32_bf16_kernel<<<(int)(E / 2048), 256, 0, stream>>>(X, xb, (int)E);
    cvt_f32_bf16_kernel<<<NW / 2048, 256, 0, stream>>>(qw, wqb, NW);
    cvt_f32_bf16_kernel<<<NW / 2048, 256, 0, stream>>>(kw, wkb, NW);
    cvt_f32_bf16_kernel<<<NW / 2048, 256, 0, stream>>>(vw, wvb, NW);
    cvt_f32_bf16_kernel<<<NW / 2048, 256, 0, stream>>>(ow, wob, NW);

    qkv_gemm_kernel<<<dim3(MROWS / 128, HIDDEN / 64, 3), 256, 0, stream>>>(
        xb, wqb, qbi, wkb, kbi, wvb, vbi, qg, kg, vg);
    attn_kernel<<<dim3(SEQ / 64, BATCH * HEADS), 256, 0, stream>>>(
        qg, kg, vg, msk, mg, Hs);
    out_gemm_kernel<<<dim3(MROWS / 128, HIDDEN / 64), 256, 0, stream>>>(
        mg, wob, obi, Hs, out);
}
